// TrainableKokoro_67181878444212
// MI455X (gfx1250) — compile-verified, matrix operand order checked by reading, not run on hardware
//
#include <hip/hip_runtime.h>
#include <hip/hip_bf16.h>
#include <math.h>

typedef _Float16 f16;
typedef __attribute__((ext_vector_type(16))) _Float16 v16h;
typedef __attribute__((ext_vector_type(8)))  _Float16 v8h;
typedef __attribute__((ext_vector_type(8)))  float    v8f;
typedef __attribute__((ext_vector_type(4)))  int      v4i;

#define T_SEQ   1024
#define NFRAMES 25600
#define HOP     600
#define DD      640     // D_MODEL + D_STY
#define HL      256     // H_LSTM
#define KDEC    672     // 642 padded to multiple of 32
#define NOPAD   640     // 600 padded to multiple of 64
#define BSTRIDE 680     // decoder LDS row stride (halves): conflict-free, 16B-mult
#define WSTRIDE 264     // k_out LDS row stride (halves): conflict-free, 16B-mult

__device__ __forceinline__ v8f wmma32(v16h a, v16h b, v8f c) {
  return __builtin_amdgcn_wmma_f32_16x16x32_f16(false, a, false, b, (short)0, c,
                                                false, false);
}

__device__ __forceinline__ v16h frag16(const f16* p) {   // p 16B-aligned
  v8h lo = *(const v8h*)p;
  v8h hi = *(const v8h*)(p + 16);
  return __builtin_shufflevector(lo, hi, 0, 1, 2, 3, 4, 5, 6, 7,
                                 8, 9, 10, 11, 12, 13, 14, 15);
}

// ------------- async global->LDS copy (ASYNCcnt path), guarded -------------

#if __has_builtin(__builtin_amdgcn_global_load_async_to_lds_b128)
typedef __attribute__((address_space(1))) v4i gv4i;
typedef __attribute__((address_space(3))) v4i lv4i;
__device__ __forceinline__ void cp_async16(const f16* g, f16* l) {
  __builtin_amdgcn_global_load_async_to_lds_b128((gv4i*)g, (lv4i*)l, 0, 0);
}
__device__ __forceinline__ void cp_wait() {
#if __has_builtin(__builtin_amdgcn_s_wait_asynccnt)
  __builtin_amdgcn_s_wait_asynccnt(0);
#else
  asm volatile("s_wait_asynccnt 0" ::: "memory");
#endif
}
#else
__device__ __forceinline__ void cp_async16(const f16* g, f16* l) {
  *(uint4*)l = *(const uint4*)g;     // synchronous fallback
}
__device__ __forceinline__ void cp_wait() {}
#endif

// ---------------- tiny prep kernels ----------------

__global__ void k_build_ids(const int* tok, int* ids) {
  int t = blockIdx.x * blockDim.x + threadIdx.x;
  if (t >= T_SEQ) return;
  int v = 0;
  if (t >= 1 && t <= 1022) { v = tok[t - 1]; if (v < 0) v = 0; }
  ids[t] = v;
}

__global__ void k_fill_style(const float* style, float* d) {
  int i = blockIdx.x * blockDim.x + threadIdx.x;          // 1024*128
  if (i >= T_SEQ * 128) return;
  int t = i >> 7, j = i & 127;
  d[t * DD + 512 + j] = style[128 + j];
}

__global__ void k_prep_wdec(const float* W_dec1, f16* AT) { // AT: 256 x KDEC
  int i = blockIdx.x * blockDim.x + threadIdx.x;
  if (i >= 256 * KDEC) return;
  int m = i / KDEC, c = i % KDEC;
  AT[i] = (c < 642) ? (f16)W_dec1[c * 256 + m] : (f16)0.f;
}

__global__ void k_prep_woutT(const float* W_out, f16* WT) { // WT: NOPAD x 256
  int i = blockIdx.x * blockDim.x + threadIdx.x;
  if (i >= NOPAD * 256) return;
  int n = i / 256, k = i % 256;
  WT[i] = (n < HOP) ? (f16)W_out[k * HOP + n] : (f16)0.f;
}

// ---------------- generic WMMA GEMM: C = act(rowgather(A) @ B + bias) -----
// M,N mult of 16; K mult of 32. One 16x16 tile per wave. act: 0=none 1=tanh
// Optional f16 mirror of C (C16/ldc16) for downstream WMMA consumers.

__global__ void k_gemm(const float* A, const float* B, float* C,
                       const int* rowids, const float* bias, f16* C16,
                       int M, int N, int K, int lda, int ldb, int ldc,
                       int ldc16, int act) {
  int lane = threadIdx.x & 31;
  int wave = threadIdx.x >> 5;
  int tiles_n = N >> 4;
  int tile = blockIdx.x * 8 + wave;                 // wave-uniform
  if (tile >= (M >> 4) * tiles_n) return;
  int m0 = (tile / tiles_n) << 4;
  int n0 = (tile % tiles_n) << 4;
  int lr = lane & 15;
  int kb = (lane >> 4) << 3;

  int arow = m0 + lr;
  if (rowids) arow = rowids[arow];
  const float* Ap = A + (long)arow * lda + kb;

  v8f acc = {};
  for (int k0 = 0; k0 < K; k0 += 32) {
    v16h a, b;
    const float* pa = Ap + k0;
    const float* pb = B + (long)(k0 + kb) * ldb + n0 + lr;
#pragma unroll
    for (int h = 0; h < 8; ++h) {
      a[h]     = (f16)pa[h];
      a[8 + h] = (f16)pa[16 + h];
      b[h]     = (f16)pb[(long)h * ldb];
      b[8 + h] = (f16)pb[(long)(16 + h) * ldb];
    }
    acc = wmma32(a, b, acc);
  }
  int col = n0 + lr;
  int r0  = m0 + ((lane >> 4) << 3);
  float bv = bias ? bias[col] : 0.f;
#pragma unroll
  for (int v = 0; v < 8; ++v) {
    float val = acc[v] + bv;
    if (act == 1) val = tanhf(val);
    C[(long)(r0 + v) * ldc + col] = val;
    if (C16) C16[(long)(r0 + v) * ldc16 + col] = (f16)val;
  }
}

// ---------------- persistent bidirectional LSTM (1 block per direction) ----

__global__ void k_lstm(const float* xg_f, const float* xg_b,
                       const float* Wh_f, const float* Wh_b, float* x) {
  __shared__ float h[HL];
  int rev = blockIdx.x;
  const float* xg = rev ? xg_b : xg_f;
  const float* Wh = rev ? Wh_b : Wh_f;
  float* xo = x + (rev ? HL : 0);
  int j = threadIdx.x;
  h[j] = 0.f;
  float cj = 0.f;
  __syncthreads();
  for (int s = 0; s < T_SEQ; ++s) {
    int t = rev ? (T_SEQ - 1 - s) : s;
    const float* g0 = xg + (long)t * 1024;
    float gi = g0[j], gf = g0[j + 256], gg = g0[j + 512], go = g0[j + 768];
    for (int i = 0; i < HL; ++i) {
      float hi = h[i];
      const float* w = Wh + (long)i * 1024 + j;
      gi += hi * w[0]; gf += hi * w[256]; gg += hi * w[512]; go += hi * w[768];
    }
    float ii = 1.f / (1.f + expf(-gi));
    float ff = 1.f / (1.f + expf(-gf));
    float oo = 1.f / (1.f + expf(-go));
    cj = ff * cj + ii * tanhf(gg);
    float hj = oo * tanhf(cj);
    xo[(long)t * 512 + j] = hj;
    __syncthreads();
    h[j] = hj;
    __syncthreads();
  }
}

// ---------------- duration head + alignment ----------------

__global__ void k_duration(const float* x, const float* W_dur,
                           const float* b_dur, int* pred) {
  int t = blockIdx.x * blockDim.x + threadIdx.x;
  if (t >= T_SEQ) return;
  const float* xt = x + (long)t * 512;
  float dur = 0.f;
  for (int n = 0; n < 50; ++n) {
    float g = b_dur[n];
    for (int k = 0; k < 512; ++k) g += xt[k] * W_dur[k * 50 + n];
    dur += 1.f / (1.f + expf(-g));
  }
  float r = rintf(dur);
  pred[t] = (int)fmaxf(r, 1.f);
}

__global__ void k_cumsum(const int* pred, int* cum) {
  if (threadIdx.x == 0 && blockIdx.x == 0) {
    int s = 0;
    for (int t = 0; t < T_SEQ; ++t) { s += pred[t]; cum[t] = s; }
  }
}

__global__ void k_indices(const int* cum, int* idx) {
  int f = blockIdx.x * blockDim.x + threadIdx.x;
  if (f >= NFRAMES) return;
  int lo = 0, hi = T_SEQ - 1;
  while (lo < hi) { int mid = (lo + hi) >> 1; if (cum[mid] > f) hi = mid; else lo = mid + 1; }
  idx[f] = lo;   // clamps to T-1 past the end (matches total_repeat_length pad)
}

__global__ void k_f0n(const float* d, const int* idx, const float* style,
                      const float* w_f0, const float* w_n, float* F0, float* Nn) {
  int f = blockIdx.x * blockDim.x + threadIdx.x;
  if (f >= NFRAMES) return;
  const float* dt = d + (long)idx[f] * DD;
  float a = 0.f, b = 0.f;
  for (int c = 0; c < DD; ++c) { float v = dt[c]; a += v * w_f0[c]; b += v * w_n[c]; }
  for (int j = 0; j < 128; ++j) {
    float v = style[128 + j];
    a += v * w_f0[DD + j]; b += v * w_n[DD + j];
  }
  F0[f] = a; Nn[f] = b;
}

// ---------------- decoder: hT[f,m] = tanh(W_dec1^T @ dec_in) ----------------
// 64 frames / block, 256 threads = 8 waves; wave w -> rows [32w,32w+32)
// dec_in staged to LDS frame-major [64][BSTRIDE] f16; t_en rows copied with
// GLOBAL_LOAD_ASYNC_TO_LDS_B128 (ASYNCcnt), extras filled by VALU.

__global__ void k_decoder(const f16* AT, const f16* t_en16, const int* idx,
                          const float* F0, const float* Nn, const float* style,
                          f16* hT) {
  extern __shared__ char smem_raw[];
  f16* Bs = (f16*)smem_raw;                         // 64 * BSTRIDE halves
  int f0 = blockIdx.x * 64;
  int tid = threadIdx.x;

  // async stage: columns 0..511 of each frame (4 threads/column, 16 x b128 each)
  {
    int fcol = tid >> 2, sub = tid & 3;
    const f16* src = t_en16 + (long)idx[f0 + fcol] * 512 + sub * 128;
    f16* dst = Bs + fcol * BSTRIDE + sub * 128;
#pragma unroll
    for (int c = 0; c < 16; ++c) cp_async16(src + c * 8, dst + c * 8);
  }
  // tail rows 512..679 (F0, N, s_ref, zero-pad)
  {
    int col2 = tid & 63, sub2 = tid >> 6;
    int ff = f0 + col2;
    for (int k = 512 + sub2 * 42; k < 512 + sub2 * 42 + 42; ++k) {
      float v;
      if (k == 512)      v = F0[ff];
      else if (k == 513) v = Nn[ff];
      else if (k < 642)  v = style[k - 514];        // s_ref
      else               v = 0.f;
      Bs[col2 * BSTRIDE + k] = (f16)v;
    }
  }
  cp_wait();
  __syncthreads();

  int lane = tid & 31;
  int wave = tid >> 5;
  int lr = lane & 15;
  int kb = (lane >> 4) << 3;
  int mbase = wave * 32;

  v8f acc[2][4] = {};
  for (int k0 = 0; k0 < KDEC; k0 += 32) {
    v16h a0 = frag16(AT + (long)(mbase + lr) * KDEC + k0 + kb);
    v16h a1 = frag16(AT + (long)(mbase + 16 + lr) * KDEC + k0 + kb);
#pragma unroll
    for (int nt = 0; nt < 4; ++nt) {
      v16h b = frag16(Bs + (nt * 16 + lr) * BSTRIDE + k0 + kb);
      acc[0][nt] = wmma32(a0, b, acc[0][nt]);
      acc[1][nt] = wmma32(a1, b, acc[1][nt]);
    }
  }
#pragma unroll
  for (int mt = 0; mt < 2; ++mt) {
#pragma unroll
    for (int nt = 0; nt < 4; ++nt) {
      int fr = f0 + nt * 16 + lr;
      int r0 = mbase + mt * 16 + ((lane >> 4) << 3);
#pragma unroll
      for (int v = 0; v < 8; ++v)
        hT[(long)fr * 256 + r0 + v] = (f16)tanhf(acc[mt][nt][v]);
    }
  }
}

// ---------------- final GEMM: audio[f,k] = hT[f,:] @ W_out[:,k] -------------
// block: 128 frames x 64 cols. W_out^T tile (64x256 f16) async-staged to LDS.

__global__ void k_out(const f16* hT, const f16* WT, float* audio) {
  extern __shared__ char smem_raw[];
  f16* Ws = (f16*)smem_raw;                         // 64 * WSTRIDE halves
  int tid = threadIdx.x;
  int nbase = blockIdx.y * 64;

  {
    int r = tid >> 2, sub = tid & 3;                // 4 threads/row, 8 x b128 each
    const f16* src = WT + (long)(nbase + r) * 256 + sub * 64;
    f16* dst = Ws + r * WSTRIDE + sub * 64;
#pragma unroll
    for (int c = 0; c < 8; ++c) cp_async16(src + c * 8, dst + c * 8);
  }
  cp_wait();
  __syncthreads();

  int lane = tid & 31;
  int wave = tid >> 5;
  int lr = lane & 15;
  int kb = (lane >> 4) << 3;
  int m0 = blockIdx.x * 128 + wave * 16;

  v8f acc[4] = {};
  const f16* Ap = hT + (long)(m0 + lr) * 256 + kb;
  for (int k0 = 0; k0 < 256; k0 += 32) {
    v16h a = frag16(Ap + k0);
#pragma unroll
    for (int nt = 0; nt < 4; ++nt) {
      v16h b = frag16(Ws + (nt * 16 + lr) * WSTRIDE + k0 + kb);
      acc[nt] = wmma32(a, b, acc[nt]);
    }
  }
  int r0 = m0 + ((lane >> 4) << 3);
#pragma unroll
  for (int nt = 0; nt < 4; ++nt) {
    int col = nbase + nt * 16 + lr;
    if (col < HOP) {
#pragma unroll
      for (int v = 0; v < 8; ++v)
        audio[(long)(r0 + v) * HOP + col] = acc[nt][v];
    }
  }
}

// ---------------- launch ----------------

extern "C" void kernel_launch(void* const* d_in, const int* in_sizes, int n_in,
                              void* d_out, int out_size, void* d_ws, size_t ws_size,
                              hipStream_t stream) {
  const int*   token_ids = (const int*)  d_in[0];
  const float* style     = (const float*)d_in[1];
  const float* emb_bert  = (const float*)d_in[2];
  const float* W_be      = (const float*)d_in[3];
  const float* W_pte     = (const float*)d_in[4];
  const float* Wi_f      = (const float*)d_in[5];
  const float* Wh_f      = (const float*)d_in[6];
  const float* b_f       = (const float*)d_in[7];
  const float* Wi_b      = (const float*)d_in[8];
  const float* Wh_b      = (const float*)d_in[9];
  const float* b_b       = (const float*)d_in[10];
  const float* W_dur     = (const float*)d_in[11];
  const float* b_dur     = (const float*)d_in[12];
  const float* w_f0      = (const float*)d_in[13];
  const float* w_n       = (const float*)d_in[14];
  const float* emb_te    = (const float*)d_in[15];
  const float* W_te      = (const float*)d_in[16];
  const float* W_dec1    = (const float*)d_in[17];
  const float* W_out     = (const float*)d_in[18];
  float* audio = (float*)d_out;

  char* ws = (char*)d_ws;
  size_t off = 0;
  auto alloc = [&](size_t bytes) {
    char* p = ws + off;
    off += (bytes + 255) & ~(size_t)255;
    return p;
  };
  int*   ids    = (int*)  alloc(T_SEQ * 4);
  float* d_en   = (float*)alloc((size_t)T_SEQ * 512 * 4);
  float* dmat   = (float*)alloc((size_t)T_SEQ * DD * 4);
  float* xg_f   = (float*)alloc((size_t)T_SEQ * 1024 * 4);
  float* xg_b   = (float*)alloc((size_t)T_SEQ * 1024 * 4);
  float* x      = (float*)alloc((size_t)T_SEQ * 512 * 4);
  int*   pred   = (int*)  alloc(T_SEQ * 4);
  int*   cum    = (int*)  alloc(T_SEQ * 4);
  int*   idx    = (int*)  alloc((size_t)NFRAMES * 4);
  float* t_en   = (float*)alloc((size_t)T_SEQ * 512 * 4);
  f16*   t_en16 = (f16*)  alloc((size_t)T_SEQ * 512 * 2);
  float* F0     = (float*)alloc((size_t)NFRAMES * 4);
  float* Nn     = (float*)alloc((size_t)NFRAMES * 4);
  f16*   AT     = (f16*)  alloc((size_t)256 * KDEC * 2);
  f16*   WT     = (f16*)  alloc((size_t)NOPAD * 256 * 2);
  f16*   hT     = (f16*)  alloc((size_t)NFRAMES * 256 * 2);

  k_build_ids<<<4, 256, 0, stream>>>(token_ids, ids);
  k_prep_wdec<<<(256 * KDEC + 255) / 256, 256, 0, stream>>>(W_dec1, AT);
  k_prep_woutT<<<(NOPAD * 256 + 255) / 256, 256, 0, stream>>>(W_out, WT);

  // d_en = emb_bert[ids] @ W_be   (1024 x 768 x 512)
  k_gemm<<<(64 * 32) / 8, 256, 0, stream>>>(emb_bert, W_be, d_en, ids, nullptr,
                                            nullptr, T_SEQ, 512, 768, 768, 512,
                                            512, 0, 0);
  // d[:, :512] = tanh(d_en @ W_pte)   (1024 x 512 x 512), ldc = 640
  k_gemm<<<(64 * 32) / 8, 256, 0, stream>>>(d_en, W_pte, dmat, nullptr, nullptr,
                                            nullptr, T_SEQ, 512, 512, 512, 512,
                                            DD, 0, 1);
  k_fill_style<<<(T_SEQ * 128) / 256, 256, 0, stream>>>(style, dmat);

  // xg = d @ Wi + b   (1024 x 640 x 1024) for both directions
  k_gemm<<<(64 * 64) / 8, 256, 0, stream>>>(dmat, Wi_f, xg_f, nullptr, b_f,
                                            nullptr, T_SEQ, 1024, DD, DD, 1024,
                                            1024, 0, 0);
  k_gemm<<<(64 * 64) / 8, 256, 0, stream>>>(dmat, Wi_b, xg_b, nullptr, b_b,
                                            nullptr, T_SEQ, 1024, DD, DD, 1024,
                                            1024, 0, 0);

  k_lstm<<<2, 256, 0, stream>>>(xg_f, xg_b, Wh_f, Wh_b, x);

  k_duration<<<4, 256, 0, stream>>>(x, W_dur, b_dur, pred);
  k_cumsum<<<1, 32, 0, stream>>>(pred, cum);
  k_indices<<<(NFRAMES + 255) / 256, 256, 0, stream>>>(cum, idx);

  k_f0n<<<(NFRAMES + 255) / 256, 256, 0, stream>>>(dmat, idx, style, w_f0, w_n,
                                                   F0, Nn);

  // t_en = tanh(emb_te[ids] @ W_te)   (1024 x 512 x 512), also f16 mirror
  k_gemm<<<(64 * 32) / 8, 256, 0, stream>>>(emb_te, W_te, t_en, ids, nullptr,
                                            t_en16, T_SEQ, 512, 512, 512, 512,
                                            512, 512, 1);

  k_decoder<<<NFRAMES / 64, 256, (size_t)64 * BSTRIDE * 2, stream>>>(
      AT, t_en16, idx, F0, Nn, style, hT);

  k_out<<<dim3(NFRAMES / 128, NOPAD / 64), 256, (size_t)64 * WSTRIDE * 2,
          stream>>>(hT, WT, audio);
}
